// Set2Set_8967891714157
// MI455X (gfx1250) — matrix operand order repaired and checked by reading, not compiled, not run on hardware
//
#include <hip/hip_runtime.h>
#include <hip/hip_bf16.h>

#define Nnodes 500000
#define Dch    128
#define Bseg   8192
#define Tsteps 6
#define Ktot   384   // 2D + D (concat q_star | h)
#define Gcols  512   // 4D gate columns

typedef __attribute__((ext_vector_type(8)))  __bf16 v8bf;
typedef __attribute__((ext_vector_type(16))) __bf16 v16bf;
typedef __attribute__((ext_vector_type(8)))  float  v8f;

__device__ __forceinline__ unsigned short f2bf(float f) {
    unsigned u = __float_as_uint(f);
    u += 0x7FFFu + ((u >> 16) & 1u);   // round-to-nearest-even
    return (unsigned short)(u >> 16);
}

__global__ void zero_f32(float* p, int n) {
    int i = blockIdx.x * blockDim.x + threadIdx.x;
    if (i < n) p[i] = 0.f;
}

// W_cat[j, 0:256] = W_ih[j,:], W_cat[j, 256:384] = W_hh[j,:]  (bf16)
__global__ void prep_W(const float* __restrict__ Wih, const float* __restrict__ Whh,
                       unsigned short* __restrict__ Wcat) {
    int i = blockIdx.x * blockDim.x + threadIdx.x;
    if (i >= Gcols * Ktot) return;
    int j = i / Ktot, k = i % Ktot;
    float v = (k < 256) ? Wih[j * 256 + k] : Whh[j * 128 + (k - 256)];
    Wcat[i] = f2bf(v);
}

// A[b, 0:256] = q_star[b,:], A[b, 256:384] = h[b,:]  (bf16)
__global__ void prep_A(const float* __restrict__ qstar, const float* __restrict__ h,
                       unsigned short* __restrict__ A) {
    int i = blockIdx.x * blockDim.x + threadIdx.x;
    if (i >= Bseg * Ktot) return;
    int b = i / Ktot, k = i % Ktot;
    float v = (k < 256) ? qstar[b * 256 + k] : h[b * 128 + (k - 256)];
    A[i] = f2bf(v);
}

// batch is sorted: start[s] = first index with batch[i] >= s ; start[B] = N
__global__ void seg_bounds(const int* __restrict__ batch, int* __restrict__ start) {
    int s = blockIdx.x * blockDim.x + threadIdx.x;
    if (s > Bseg) return;
    if (s == Bseg) { start[Bseg] = Nnodes; return; }
    int lo = 0, hi = Nnodes;
    while (lo < hi) { int mid = (lo + hi) >> 1; if (batch[mid] < s) lo = mid + 1; else hi = mid; }
    start[s] = lo;
}

// gates[8192,512] = A_bf16[8192,384] @ W_cat_bf16[512,384]^T via v_wmma_f32_16x16x32_bf16
// block = 8 waves (256 thr); block covers 16 rows x 512 cols; wave w covers cols [w*64, w*64+64)
// K loop software-pipelined: next A/B fragments load while current WMMAs execute.
__global__ __launch_bounds__(256) void gemm_gates(const unsigned short* __restrict__ A_,
                                                  const unsigned short* __restrict__ W_,
                                                  float* __restrict__ gates) {
    const __bf16* A = (const __bf16*)A_;
    const __bf16* W = (const __bf16*)W_;
    const int row0 = blockIdx.x * 16;
    const int wave = threadIdx.x >> 5;
    const int lane = threadIdx.x & 31;
    const int lm   = lane & 15;   // M (A) / N (B) / N (C) index
    const int lh   = lane >> 4;   // K-half (A/B), row-half (C)

    v8f acc[4];
#pragma unroll
    for (int j = 0; j < 4; ++j) acc[j] = (v8f){0.f,0.f,0.f,0.f,0.f,0.f,0.f,0.f};

    // A frag source: row (row0+lm), K = {kk+lh*8..+7, kk+16+lh*8..+7}
    const __bf16* arow = A + (size_t)(row0 + lm) * Ktot + lh * 8;
    // B frag source: W row (col0+lm), K = {kk+lh*8..+7, kk+16+lh*8..+7}
    const __bf16* wrow[4];
#pragma unroll
    for (int j = 0; j < 4; ++j)
        wrow[j] = W + (size_t)((wave * 4 + j) * 16 + lm) * Ktot + lh * 8;

    // prologue: load fragments for kk = 0
    v16bf af, bf[4];
    {
        v8bf a0 = *(const v8bf*)(arow);
        v8bf a1 = *(const v8bf*)(arow + 16);
        af = __builtin_shufflevector(a0, a1, 0,1,2,3,4,5,6,7,8,9,10,11,12,13,14,15);
#pragma unroll
        for (int j = 0; j < 4; ++j) {
            v8bf b0 = *(const v8bf*)(wrow[j]);
            v8bf b1 = *(const v8bf*)(wrow[j] + 16);
            bf[j] = __builtin_shufflevector(b0, b1, 0,1,2,3,4,5,6,7,8,9,10,11,12,13,14,15);
        }
    }

#pragma unroll
    for (int kk = 0; kk < Ktot; kk += 32) {
        v16bf afn, bfn[4];
        if (kk + 32 < Ktot) {   // prefetch next K-step into disjoint registers
            v8bf a0 = *(const v8bf*)(arow + kk + 32);
            v8bf a1 = *(const v8bf*)(arow + kk + 48);
            afn = __builtin_shufflevector(a0, a1, 0,1,2,3,4,5,6,7,8,9,10,11,12,13,14,15);
#pragma unroll
            for (int j = 0; j < 4; ++j) {
                v8bf b0 = *(const v8bf*)(wrow[j] + kk + 32);
                v8bf b1 = *(const v8bf*)(wrow[j] + kk + 48);
                bfn[j] = __builtin_shufflevector(b0, b1, 0,1,2,3,4,5,6,7,8,9,10,11,12,13,14,15);
            }
        }
#pragma unroll
        for (int j = 0; j < 4; ++j)
            acc[j] = __builtin_amdgcn_wmma_f32_16x16x32_bf16(
                false, af, false, bf[j], (short)0, acc[j], false, false);
        if (kk + 32 < Ktot) {
            af = afn;
#pragma unroll
            for (int j = 0; j < 4; ++j) bf[j] = bfn[j];
        }
    }

    // C/D layout: element (m,n): lane = (n%16) + 16*(m>=8), VGPR r = m%8
#pragma unroll
    for (int j = 0; j < 4; ++j) {
        int col = (wave * 4 + j) * 16 + lm;
#pragma unroll
        for (int r = 0; r < 8; ++r) {
            int m = r + 8 * lh;
            gates[(size_t)(row0 + m) * Gcols + col] = acc[j][r];
        }
    }
}

__global__ void lstm_update(const float* __restrict__ gates,
                            const float* __restrict__ bih, const float* __restrict__ bhh,
                            float* __restrict__ c, float* __restrict__ h) {
    int i = blockIdx.x * blockDim.x + threadIdx.x;
    if (i >= Bseg * Dch) return;
    int b = i / Dch, d = i % Dch;
    const float* g = gates + (size_t)b * Gcols;
    float gi = g[d      ] + bih[d      ] + bhh[d      ];
    float gf = g[128 + d] + bih[128 + d] + bhh[128 + d];
    float gg = g[256 + d] + bih[256 + d] + bhh[256 + d];
    float go = g[384 + d] + bih[384 + d] + bhh[384 + d];
    float ii = 1.f / (1.f + __expf(-gi));
    float ff = 1.f / (1.f + __expf(-gf));
    float tg = tanhf(gg);
    float oo = 1.f / (1.f + __expf(-go));
    float cn = ff * c[i] + ii * tg;
    c[i] = cn;
    h[i] = oo * tanhf(cn);
}

// One wave per segment (batch sorted -> contiguous). No atomics.
// loop1: e[n] = <x[n], h[seg]>, running max. loop2: exp + denom. loop3: r = sum a*x.
__global__ __launch_bounds__(256) void attn(const float* __restrict__ x,
                                            const int* __restrict__ start,
                                            const float* __restrict__ h,
                                            float* __restrict__ e,
                                            float* __restrict__ qout) {
    int wid  = blockIdx.x * 8 + (threadIdx.x >> 5);
    int lane = threadIdx.x & 31;
    if (wid >= Bseg) return;
    int s0 = start[wid], s1 = start[wid + 1];
    const float4* xp = (const float4*)x;
    float4 q = ((const float4*)(h + (size_t)wid * Dch))[lane];

    float mx = -3.402823466e38f;
    for (int n = s0; n < s1; ++n) {
        float4 xv = xp[(size_t)n * 32 + lane];
        float v = q.x * xv.x + q.y * xv.y + q.z * xv.z + q.w * xv.w;
#pragma unroll
        for (int m = 16; m >= 1; m >>= 1) v += __shfl_xor(v, m, 32);
        if (lane == 0) e[n] = v;
        mx = fmaxf(mx, v);
    }
    __threadfence();
    float ds = 0.f;
    for (int n = s0 + lane; n < s1; n += 32) {
        float ex = __expf(e[n] - mx);
        e[n] = ex;
        ds += ex;
    }
#pragma unroll
    for (int m = 16; m >= 1; m >>= 1) ds += __shfl_xor(ds, m, 32);
    __threadfence();
    float inv = 1.f / (ds + 1e-16f);
    float rx = 0.f, ry = 0.f, rz = 0.f, rw = 0.f;
    for (int n = s0; n < s1; ++n) {
        float a = e[n] * inv;
        float4 xv = xp[(size_t)n * 32 + lane];
        rx += a * xv.x; ry += a * xv.y; rz += a * xv.z; rw += a * xv.w;
    }
    float4* qo = (float4*)(qout + (size_t)wid * 256);
    qo[lane] = q;                              // q_star[:,0:128]   = h
    float4 rr; rr.x = rx; rr.y = ry; rr.z = rz; rr.w = rw;
    qo[32 + lane] = rr;                        // q_star[:,128:256] = r
}

extern "C" void kernel_launch(void* const* d_in, const int* in_sizes, int n_in,
                              void* d_out, int out_size, void* d_ws, size_t ws_size,
                              hipStream_t stream) {
    const float* x     = (const float*)d_in[0];
    const int*   batch = (const int*)d_in[1];
    const float* Wih   = (const float*)d_in[2];
    const float* Whh   = (const float*)d_in[3];
    const float* bih   = (const float*)d_in[4];
    const float* bhh   = (const float*)d_in[5];

    char* ws = (char*)d_ws;
    size_t off = 0;
    auto salloc = [&](size_t bytes) -> char* {
        char* p = ws + off;
        off = (off + bytes + 255) & ~(size_t)255;
        return p;
    };
    float*          h     = (float*)salloc((size_t)Bseg * Dch * 4);
    float*          c     = (float*)salloc((size_t)Bseg * Dch * 4);
    float*          qstar = (float*)salloc((size_t)Bseg * 256 * 4);
    float*          gates = (float*)salloc((size_t)Bseg * Gcols * 4);
    float*          e     = (float*)salloc((size_t)Nnodes * 4);
    unsigned short* Abf   = (unsigned short*)salloc((size_t)Bseg * Ktot * 2);
    unsigned short* Wcat  = (unsigned short*)salloc((size_t)Gcols * Ktot * 2);
    int*            start = (int*)salloc((size_t)(Bseg + 1) * 4);

    zero_f32<<<(Bseg * Dch + 255) / 256, 256, 0, stream>>>(h, Bseg * Dch);
    zero_f32<<<(Bseg * Dch + 255) / 256, 256, 0, stream>>>(c, Bseg * Dch);
    zero_f32<<<(Bseg * 256 + 255) / 256, 256, 0, stream>>>(qstar, Bseg * 256);
    prep_W<<<(Gcols * Ktot + 255) / 256, 256, 0, stream>>>(Wih, Whh, Wcat);
    seg_bounds<<<(Bseg + 1 + 255) / 256, 256, 0, stream>>>(batch, start);

    for (int t = 0; t < Tsteps; ++t) {
        prep_A<<<(Bseg * Ktot + 255) / 256, 256, 0, stream>>>(qstar, h, Abf);
        gemm_gates<<<Bseg / 16, 256, 0, stream>>>(Abf, Wcat, gates);
        lstm_update<<<(Bseg * Dch + 255) / 256, 256, 0, stream>>>(gates, bih, bhh, c, h);
        float* qo = (t == Tsteps - 1) ? (float*)d_out : qstar;
        attn<<<Bseg / 8, 256, 0, stream>>>(x, start, h, e, qo);
    }
}
